// SECONDMambaBlock_49005576847551
// MI455X (gfx1250) — matrix operand order repaired and checked by reading, not compiled, not run on hardware
//
#include <hip/hip_runtime.h>
#include <hip/hip_bf16.h>

// ---------------- problem constants ----------------
#define DM      256
#define DSTATE  4
#define HDIM    64
#define NH      4
#define HH      180
#define WWID    180
#define HW      (HH*WWID)      // 32400
#define LTOK    HW
#define NCHUNK  127
#define LP      (NCHUNK*256)   // 32512 (padded length)
#define DPROJ   524
#define DPROJP  528            // Zx row pitch
#define WROWSP  576            // padded in-proj weight rows (36*16)
#define LN_EPS  1e-6f

typedef __attribute__((ext_vector_type(16))) __bf16 v16bf;
typedef __attribute__((ext_vector_type(8)))  __bf16 v8bf;
typedef __attribute__((ext_vector_type(8)))  float  v8f;
typedef unsigned int u32x4 __attribute__((ext_vector_type(4)));
typedef int          i32x8 __attribute__((ext_vector_type(8)));
typedef int          i32x4 __attribute__((ext_vector_type(4)));

#if __has_builtin(__builtin_amdgcn_tensor_load_to_lds) && \
    __has_builtin(__builtin_amdgcn_s_wait_tensorcnt)
#define USE_TDM 1
#else
#define USE_TDM 0
#endif

static __device__ __forceinline__ v16bf cat8(v8bf a, v8bf b) {
  v16bf r;
#pragma unroll
  for (int i = 0; i < 8; ++i) { r[i] = a[i]; r[i + 8] = b[i]; }
  return r;
}
static __device__ __forceinline__ float siluf(float x) {
  return x / (1.0f + __expf(-x));
}
static __device__ __forceinline__ float softplusf(float x) {
  return (x > 20.0f) ? x : log1pf(__expf(x));
}

#if USE_TDM
// Issue a TDM 2-D tile load (bf16 elements) global -> LDS, with LDS row
// padding of 16B per 64B row (reproduces the bank-stagger pitch APAD=40).
static __device__ __forceinline__ void tdm_tile_2d(unsigned ldsAddr, const void* g,
                                                   int tileW, int tileH,
                                                   int strideElems) {
  unsigned long long ga = (unsigned long long)g;
  u32x4 g0;
  g0.x = 1u;                                                // count=1 (valid user D#)
  g0.y = ldsAddr;                                           // LDS byte address
  g0.z = (unsigned)ga;                                      // global_addr[31:0]
  g0.w = (unsigned)((ga >> 32) & 0x01ffffffu) | (2u << 30); // ga[56:32] | type=2
  const unsigned TD = 1u << 20;  // huge tensor dims: tiles never clipped OOB
  i32x8 g1;
  g1[0] = (1 << 16)     // data_size = 2 bytes
        | (1 << 20)     // pad_enable
        | (3 << 22)     // pad_interval: 16 DWORDs (= one 64B tile row)
        | (3 << 25);    // pad_amount: 4 DWORDs (16B) -> LDS pitch 80B
  g1[1] = (int)((TD & 0xffffu) << 16);                          // td0[15:0]
  g1[2] = (int)(((TD >> 16) & 0xffffu) | ((TD & 0xffffu) << 16)); // td0 hi | td1 lo
  g1[3] = (int)(((TD >> 16) & 0xffffu) | ((unsigned)tileW << 16)); // td1 hi | tile_dim0
  g1[4] = tileH & 0xffff;                                       // tile_dim1 (dim2=0)
  g1[5] = strideElems;                                          // tensor_dim0_stride lo
  g1[6] = 0;
  g1[7] = 0;
  i32x4 z4 = {0, 0, 0, 0};
  i32x8 z8 = {0, 0, 0, 0, 0, 0, 0, 0};
  __builtin_amdgcn_tensor_load_to_lds(g0, g1, z4, z4, z8, 0);
}
#endif

// ---------------- K0: fp32 -> bf16 weight convert with zero row padding ----
__global__ void k_convert_pad(const float* __restrict__ src, __bf16* __restrict__ dst,
                              int srcRows, int dstRows, int cols) {
  int id = blockIdx.x * 256 + threadIdx.x;
  if (id >= dstRows * cols) return;
  int r = id / cols;
  float v = (r < srcRows) ? src[id] : 0.0f;
  dst[id] = (__bf16)v;
}

// ---------------- K1: Morton gather + transpose -> bf16 activation matrix --
// Xm[l][c] = x[c][morton[l]]  (writes coalesced; gather reads served by L2)
__global__ void k_gather(const float* __restrict__ x, const int* __restrict__ mort,
                         __bf16* __restrict__ Xm) {
  int id = blockIdx.x * 256 + threadIdx.x;   // over LP*256
  int l = id >> 8, c = id & 255;
  float v = 0.0f;
  if (l < LTOK) v = x[c * HW + mort[l]];
  Xm[id] = (__bf16)v;
}

// ---------------- K2: in-projection GEMM  Zx = Xm(LP x 256) @ W^T ----------
// BM=64, BN=64, BK=32; 8 waves: wave=(mw,nw), each wave 2 Mtiles x 1 Ntile.
// Double-buffered TDM (tensor_load_to_lds) stages A/B tiles straight to LDS.
#define APAD 40
__global__ __launch_bounds__(256) void k_gemm_in(const __bf16* __restrict__ Xm,
                                                 const __bf16* __restrict__ Wb,
                                                 float* __restrict__ Zx) {
  __shared__ __bf16 aT[2][64][APAD];
  __shared__ __bf16 bT[2][64][APAD];
  const int m0 = blockIdx.x * 64, n0 = blockIdx.y * 64;
  const int t = threadIdx.x;
  const int w = t >> 5, lane = t & 31, half = lane >> 4, m = lane & 15;
  const int mw = w & 1, nw = w >> 1;
  v8f acc0 = {};
  v8f acc1 = {};
  int buf = 0;
#if USE_TDM
  if (t < 32) {   // wave 0 drives the Tensor Data Mover
    tdm_tile_2d((unsigned)(unsigned long long)&aT[0][0][0], &Xm[m0 * 256], 32, 64, 256);
    tdm_tile_2d((unsigned)(unsigned long long)&bT[0][0][0], &Wb[n0 * 256], 32, 64, 256);
  }
#else
  const int r = t >> 2, seg = t & 3;
#endif
  for (int k0 = 0; k0 < 256; k0 += 32) {
#if USE_TDM
    if (t < 32) {
      if (k0 + 32 < 256) {   // kick next slice, then wait for current (in-order)
        tdm_tile_2d((unsigned)(unsigned long long)&aT[buf ^ 1][0][0],
                    &Xm[m0 * 256 + k0 + 32], 32, 64, 256);
        tdm_tile_2d((unsigned)(unsigned long long)&bT[buf ^ 1][0][0],
                    &Wb[n0 * 256 + k0 + 32], 32, 64, 256);
        __builtin_amdgcn_s_wait_tensorcnt((short)2);
      } else {
        __builtin_amdgcn_s_wait_tensorcnt((short)0);
      }
    }
    __syncthreads();
#else
    __syncthreads();
    *(v8bf*)&aT[0][r][seg * 8] = *(const v8bf*)&Xm[(m0 + r) * 256 + k0 + seg * 8];
    *(v8bf*)&bT[0][r][seg * 8] = *(const v8bf*)&Wb[(n0 + r) * 256 + k0 + seg * 8];
    if (k0 + 32 < 256) {
      __builtin_prefetch(&Xm[(m0 + r) * 256 + k0 + 32 + seg * 8], 0, 1);
      __builtin_prefetch(&Wb[(n0 + r) * 256 + k0 + 32 + seg * 8], 0, 1);
    }
    __syncthreads();
#endif
    const __bf16 (*aTb)[APAD] = aT[buf];
    const __bf16 (*bTb)[APAD] = bT[buf];
    const __bf16* br = &bTb[nw * 16 + m][half * 16];
    v16bf B = cat8(*(const v8bf*)br, *(const v8bf*)(br + 8));
    const __bf16* ar0 = &aTb[(2 * mw + 0) * 16 + m][0];
    v16bf A0 = cat8(*(const v8bf*)(ar0 + half * 16), *(const v8bf*)(ar0 + half * 16 + 8));
    acc0 = __builtin_amdgcn_wmma_f32_16x16x32_bf16(false, A0, false, B, (short)0, acc0, false, false);
    const __bf16* ar1 = &aTb[(2 * mw + 1) * 16 + m][0];
    v16bf A1 = cat8(*(const v8bf*)(ar1 + half * 16), *(const v8bf*)(ar1 + half * 16 + 8));
    acc1 = __builtin_amdgcn_wmma_f32_16x16x32_bf16(false, A1, false, B, (short)0, acc1, false, false);
#if USE_TDM
    buf ^= 1;
    __syncthreads();   // readers done before the in-flight TDM's buffer reuse
#endif
  }
  const int ncol = n0 + nw * 16 + m;
  if (ncol < DPROJP) {
#pragma unroll
    for (int j = 0; j < 8; ++j) {
      int row0 = m0 + (2 * mw + 0) * 16 + j + 8 * half;
      int row1 = m0 + (2 * mw + 1) * 16 + j + 8 * half;
      Zx[(long)row0 * DPROJP + ncol] = acc0[j];
      Zx[(long)row1 * DPROJP + ncol] = acc1[j];
    }
  }
}

// ---------------- K3a: xs = silu(x-part of Zx) as bf16 ---------------------
__global__ void k_xs(const float* __restrict__ Zx, __bf16* __restrict__ xsB) {
  int id = blockIdx.x * 256 + threadIdx.x;   // over LP*256
  int l = id >> 8, c = id & 255;
  xsB[id] = (__bf16)siluf(Zx[(long)l * DPROJP + 256 + c]);
}

// ---------------- K3b: B/C/dt prep + per-chunk cumsum of a=dt*A ------------
__global__ __launch_bounds__(256) void k_prep(const float* __restrict__ Zx,
                                              const float* __restrict__ dt_bias,
                                              const float* __restrict__ A_log,
                                              float* __restrict__ Bn, float* __restrict__ Cn,
                                              float* __restrict__ dtv,
                                              float* __restrict__ acs, float* __restrict__ csum) {
  __shared__ float sc[256];
  const int c = blockIdx.x, t = threadIdx.x;
  const int gl = c * 256 + t;
  const float* row = &Zx[(long)gl * DPROJP];
#pragma unroll
  for (int n = 0; n < 4; ++n) {
    Bn[gl * 4 + n] = siluf(row[512 + n]);
    Cn[gl * 4 + n] = siluf(row[516 + n]);
  }
  for (int h = 0; h < NH; ++h) {
    float dt = softplusf(row[520 + h] + dt_bias[h]);
    float a = dt * (-__expf(A_log[h]));
    dtv[gl * 4 + h] = dt;
    sc[t] = a;
    __syncthreads();
    for (int off = 1; off < 256; off <<= 1) {   // Hillis-Steele inclusive scan
      float v = (t >= off) ? sc[t - off] : 0.0f;
      __syncthreads();
      sc[t] += v;
      __syncthreads();
    }
    acs[gl * 4 + h] = sc[t];
    if (t == 255) csum[c * 4 + h] = sc[255];
    __syncthreads();
  }
}

// ---------------- K4: per-chunk end states ---------------------------------
// states[c,h,p,n] = sum_l B[l,n] * exp(tot - acs[l]) * dt[l] * silu_x[l,h*64+p]
__global__ __launch_bounds__(256) void k_states(const __bf16* __restrict__ xsB,
                                                const float* __restrict__ Bn,
                                                const float* __restrict__ dtv,
                                                const float* __restrict__ acs,
                                                const float* __restrict__ csum,
                                                float* __restrict__ states) {
  const int c = blockIdx.x, h = blockIdx.y, t = threadIdx.x;
  const int p = t >> 2, n = t & 3;
  const float tot = csum[c * 4 + h];
  float acc = 0.0f;
  const int gl0 = c * 256;
  for (int l = 0; l < 256; ++l) {
    int gl = gl0 + l;
    float xv = (float)xsB[(long)gl * 256 + h * 64 + p];
    acc += Bn[gl * 4 + n] * __expf(tot - acs[gl * 4 + h]) * dtv[gl * 4 + h] * xv;
  }
  states[(c * NH + h) * 256 + t] = acc;
}

// ---------------- K5: sequential inter-chunk recurrence --------------------
__global__ void k_chunkscan(const float* __restrict__ states, const float* __restrict__ csum,
                            float* __restrict__ prev) {
  const int h = blockIdx.x, t = threadIdx.x;   // t = p*4+n
  float run = 0.0f;
  for (int c = 0; c < NCHUNK; ++c) {
    int idx = (c * NH + h) * 256 + t;
    prev[idx] = run;
    run = run * __expf(csum[c * 4 + h]) + states[idx];
  }
}

// ---------------- K6: per-(chunk,head) Y = G@X (WMMA) + Y_off + gating -----
#define XP 264
#define GP 272
__global__ __launch_bounds__(256) void k_ydiag(const float* __restrict__ Zx,
                                               const __bf16* __restrict__ xsB,
                                               const float* __restrict__ Bn,
                                               const float* __restrict__ Cn,
                                               const float* __restrict__ dtv,
                                               const float* __restrict__ acs,
                                               const float* __restrict__ prev,
                                               const float* __restrict__ Darr,
                                               float* __restrict__ Y) {
  __shared__ __bf16 XtS[64][XP];     // X^T : [p][s], X = dt*silu(x)
  __shared__ __bf16 GS[32][GP];      // 32-row slab of G
  __shared__ float Cs[256][4];
  __shared__ float BsS[256][4];
  __shared__ float acsS[256];
  __shared__ float prevS[64][4];
  const int c = blockIdx.x, h = blockIdx.y;
  const int t = threadIdx.x;
  const int gl0 = c * 256;
  {
    const int s = t, gl = gl0 + s;
#pragma unroll
    for (int n = 0; n < 4; ++n) { BsS[s][n] = Bn[gl * 4 + n]; Cs[s][n] = Cn[gl * 4 + n]; }
    acsS[s] = acs[gl * 4 + h];
    float dts = dtv[gl * 4 + h];
    const __bf16* xr = &xsB[(long)gl * 256 + h * 64];
    for (int p = 0; p < 64; ++p) XtS[p][s] = (__bf16)((float)xr[p] * dts);
    prevS[t >> 2][t & 3] = prev[(c * NH + h) * 256 + t];
  }
  __syncthreads();
  const float Dh = Darr[h];
  const int w = t >> 5, lane = t & 31, half = lane >> 4, m = lane & 15;
  const int mt = w & 1, nt = w >> 1;
  for (int rb = 0; rb < 8; ++rb) {
    const int r0 = rb * 32;
    {  // build G slab: thread t owns column s=t
      const int s = t;
      const float b0 = BsS[s][0], b1 = BsS[s][1], b2 = BsS[s][2], b3 = BsS[s][3];
      const float as_ = acsS[s];
#pragma unroll
      for (int li = 0; li < 32; ++li) {
        int l = r0 + li;
        float g = 0.0f;
        if (s <= l)
          g = (Cs[l][0] * b0 + Cs[l][1] * b1 + Cs[l][2] * b2 + Cs[l][3] * b3) *
              __expf(acsS[l] - as_);
        GS[li][s] = (__bf16)g;
      }
    }
    __syncthreads();
    v8f acc = {};
    for (int k0 = 0; k0 < 256; k0 += 32) {
      const __bf16* ar = &GS[mt * 16 + m][k0];
      v16bf A = cat8(*(const v8bf*)(ar + half * 16), *(const v8bf*)(ar + half * 16 + 8));
      const __bf16* br = &XtS[nt * 16 + m][k0 + half * 16];
      v16bf B = cat8(*(const v8bf*)br, *(const v8bf*)(br + 8));
      acc = __builtin_amdgcn_wmma_f32_16x16x32_bf16(false, A, false, B, (short)0, acc, false, false);
    }
    const int p = nt * 16 + m;
    const int pg = h * 64 + p;
#pragma unroll
    for (int j = 0; j < 8; ++j) {
      int l = r0 + mt * 16 + j + 8 * half;
      int gl = gl0 + l;
      float yoff = __expf(acsS[l]) * (Cs[l][0] * prevS[p][0] + Cs[l][1] * prevS[p][1] +
                                      Cs[l][2] * prevS[p][2] + Cs[l][3] * prevS[p][3]);
      float xv = siluf(Zx[(long)gl * DPROJP + 256 + pg]);
      float zv = Zx[(long)gl * DPROJP + pg];
      Y[(long)gl * 256 + pg] = (acc[j] + yoff + Dh * xv) * siluf(zv);
    }
    __syncthreads();
  }
}

// ---------------- K7: fused un-Morton gather + out-proj (WMMA) + LayerNorm -
#define CP  536
#define OP2 264
__global__ __launch_bounds__(256) void k_outproj_ln(const float* __restrict__ YH,
                                                    const float* __restrict__ YV,
                                                    const int* __restrict__ invH,
                                                    const int* __restrict__ invV,
                                                    const __bf16* __restrict__ Wo,
                                                    const float* __restrict__ gamma,
                                                    const float* __restrict__ beta,
                                                    float* __restrict__ out) {
  __shared__ __bf16 catL[16][CP];    // gathered [YH | YV] rows, bf16
  __shared__ float outL[16][OP2];
  __shared__ float gS[256], bS[256], muS[16], rsS[16];
  const int l0 = blockIdx.x * 16;
  const int t = threadIdx.x;
  gS[t] = gamma[t];
  bS[t] = beta[t];
  for (int idx = t; idx < 16 * 512; idx += 256) {
    int r = idx >> 9, cc = idx & 511;
    int l = l0 + r;
    float v = (cc < 256) ? YH[(long)invH[l] * 256 + cc]
                         : YV[(long)invV[l] * 256 + (cc - 256)];
    catL[r][cc] = (__bf16)v;
  }
  __syncthreads();
  const int w = t >> 5, lane = t & 31, half = lane >> 4, m = lane & 15;
  const int n0 = w * 32;
  v8f acc0 = {};
  v8f acc1 = {};
  for (int k0 = 0; k0 < 512; k0 += 32) {
    const __bf16* ar = &catL[m][k0];
    v16bf A = cat8(*(const v8bf*)(ar + half * 16), *(const v8bf*)(ar + half * 16 + 8));
    const __bf16* b0p = &Wo[(n0 + m) * 512 + k0 + half * 16];
    v16bf B0 = cat8(*(const v8bf*)b0p, *(const v8bf*)(b0p + 8));
    acc0 = __builtin_amdgcn_wmma_f32_16x16x32_bf16(false, A, false, B0, (short)0, acc0, false, false);
    const __bf16* b1p = &Wo[(n0 + 16 + m) * 512 + k0 + half * 16];
    v16bf B1 = cat8(*(const v8bf*)b1p, *(const v8bf*)(b1p + 8));
    acc1 = __builtin_amdgcn_wmma_f32_16x16x32_bf16(false, A, false, B1, (short)0, acc1, false, false);
  }
#pragma unroll
  for (int j = 0; j < 8; ++j) {
    int r = j + 8 * half;
    outL[r][n0 + m] = acc0[j];
    outL[r][n0 + 16 + m] = acc1[j];
  }
  __syncthreads();
  if (t < 16) {
    float s = 0.0f, s2 = 0.0f;
    for (int mm = 0; mm < 256; ++mm) { float v = outL[t][mm]; s += v; s2 += v * v; }
    float mu = s * (1.0f / 256.0f);
    float var = s2 * (1.0f / 256.0f) - mu * mu;
    muS[t] = mu;
    rsS[t] = rsqrtf(var + LN_EPS);
  }
  __syncthreads();
  for (int idx = t; idx < 4096; idx += 256) {
    int r = idx & 15, mm = idx >> 4;
    float v = (outL[r][mm] - muS[r]) * rsS[r] * gS[mm] + bS[mm];
    out[(long)mm * HW + l0 + r] = v;   // (1,256,H,W) layout
  }
}

// ---------------- host launcher --------------------------------------------
extern "C" void kernel_launch(void* const* d_in, const int* in_sizes, int n_in,
                              void* d_out, int out_size, void* d_ws, size_t ws_size,
                              hipStream_t stream) {
  const float* x     = (const float*)d_in[0];
  const float* WinH  = (const float*)d_in[1];
  const float* WinV  = (const float*)d_in[2];
  const float* dtbH  = (const float*)d_in[3];
  const float* dtbV  = (const float*)d_in[4];
  const float* AlogH = (const float*)d_in[5];
  const float* AlogV = (const float*)d_in[6];
  const float* DHa   = (const float*)d_in[7];
  const float* DVa   = (const float*)d_in[8];
  const float* Wout  = (const float*)d_in[9];
  const float* gamma = (const float*)d_in[10];
  const float* beta  = (const float*)d_in[11];
  const int* mort[2] = { (const int*)d_in[12], (const int*)d_in[13] };
  const int* invH    = (const int*)d_in[14];
  const int* invV    = (const int*)d_in[15];
  float* out = (float*)d_out;
  (void)in_sizes; (void)n_in; (void)out_size; (void)ws_size;

  char* ws = (char*)d_ws;
  size_t off = 0;
  auto alloc = [&](size_t bytes) -> char* {
    char* p = ws + off;
    off = (off + bytes + 255) & ~(size_t)255;
    return p;
  };
  __bf16* wInBf[2] = { (__bf16*)alloc((size_t)WROWSP * 256 * 2),
                       (__bf16*)alloc((size_t)WROWSP * 256 * 2) };
  __bf16* wOutBf = (__bf16*)alloc((size_t)256 * 512 * 2);
  __bf16* Xm[2]; float* Zx[2]; __bf16* xsB[2];
  float *Bn[2], *Cn[2], *dtv[2], *acsB[2], *csum[2], *states[2], *prevB[2], *Yb[2];
  for (int p = 0; p < 2; ++p) {
    Xm[p]    = (__bf16*)alloc((size_t)LP * 256 * 2);
    Zx[p]    = (float*)alloc((size_t)LP * DPROJP * 4);
    xsB[p]   = (__bf16*)alloc((size_t)LP * 256 * 2);
    Bn[p]    = (float*)alloc((size_t)LP * 4 * 4);
    Cn[p]    = (float*)alloc((size_t)LP * 4 * 4);
    dtv[p]   = (float*)alloc((size_t)LP * 4 * 4);
    acsB[p]  = (float*)alloc((size_t)LP * 4 * 4);
    csum[p]  = (float*)alloc((size_t)NCHUNK * 4 * 4);
    states[p] = (float*)alloc((size_t)NCHUNK * NH * HDIM * DSTATE * 4);
    prevB[p]  = (float*)alloc((size_t)NCHUNK * NH * HDIM * DSTATE * 4);
    Yb[p]    = (float*)alloc((size_t)LP * 256 * 4);
  }
  const float* dtb[2]  = { dtbH, dtbV };
  const float* Alog[2] = { AlogH, AlogV };
  const float* Darr[2] = { DHa, DVa };

  // K0: weights -> bf16 (zero-padded rows for in-proj)
  int nconv = WROWSP * 256;
  k_convert_pad<<<(nconv + 255) / 256, 256, 0, stream>>>(WinH, wInBf[0], DPROJ, WROWSP, 256);
  k_convert_pad<<<(nconv + 255) / 256, 256, 0, stream>>>(WinV, wInBf[1], DPROJ, WROWSP, 256);
  k_convert_pad<<<(256 * 512 + 255) / 256, 256, 0, stream>>>(Wout, wOutBf, 256, 256, 512);

  for (int p = 0; p < 2; ++p) {
    // K1: Morton gather (bf16, zero tail)
    k_gather<<<LP, 256, 0, stream>>>(x, mort[p], Xm[p]);
    // K2: in-projection GEMM via WMMA + TDM double-buffered staging
    dim3 g2(LP / 64, WROWSP / 64);
    k_gemm_in<<<g2, 256, 0, stream>>>(Xm[p], wInBf[p], Zx[p]);
    // K3: activations + per-chunk scan of a = dt*A
    k_xs<<<LP, 256, 0, stream>>>(Zx[p], xsB[p]);
    k_prep<<<NCHUNK, 256, 0, stream>>>(Zx[p], dtb[p], Alog[p], Bn[p], Cn[p], dtv[p],
                                       acsB[p], csum[p]);
    // K4: chunk end-states
    dim3 g4(NCHUNK, NH);
    k_states<<<g4, 256, 0, stream>>>(xsB[p], Bn[p], dtv[p], acsB[p], csum[p], states[p]);
    // K5: sequential inter-chunk recurrence (tiny)
    k_chunkscan<<<NH, 256, 0, stream>>>(states[p], csum[p], prevB[p]);
    // K6: Y = (G @ X) + Y_off + D*x, gated by silu(z)  (WMMA)
    k_ydiag<<<g4, 256, 0, stream>>>(Zx[p], xsB[p], Bn[p], Cn[p], dtv[p], acsB[p],
                                    prevB[p], Darr[p], Yb[p]);
  }
  // K7: un-Morton gather + out-projection (WMMA) + LayerNorm + transpose store
  k_outproj_ln<<<HW / 16, 256, 0, stream>>>(Yb[0], Yb[1], invH, invV, wOutBf,
                                            gamma, beta, out);
}
